// GATNetwork_9337258902051
// MI455X (gfx1250) — matrix operand order repaired and checked by reading, not compiled, not run on hardware
//
#include <hip/hip_runtime.h>
#include <hip/hip_bf16.h>

// ---------------- problem constants (match reference) ----------------
#define NN     50000
#define BB     100
#define NPGG   500
#define EORIG  800000
#define ETOT   850000     // EORIG + NN self loops
#define HH     4
#define CC     32
#define HCC    128
#define EDD    8
#define AGGD   192
#define OUTD   8

typedef __attribute__((ext_vector_type(16))) _Float16 v16h;
typedef __attribute__((ext_vector_type(8)))  float    v8f;

// ---------------- small helpers ----------------
__global__ void fill_f32(float* __restrict__ p, float v, int n) {
  int i = blockIdx.x * blockDim.x + threadIdx.x;
  if (i < n) p[i] = v;
}

__device__ __forceinline__ void atomicMaxF32(float* addr, float v) {
  if (v >= 0.0f) atomicMax((int*)addr, __float_as_int(v));
  else           atomicMin((unsigned int*)addr, __float_as_uint(v));
}

// ---------------- self-loop edge-attr (fill_value='mean') ----------------
__global__ void sl_accum(const float* __restrict__ ea, const int* __restrict__ dst,
                         float* __restrict__ sl, float* __restrict__ deg) {
  int e = blockIdx.x * blockDim.x + threadIdx.x;
  if (e >= EORIG) return;
  int d = dst[e];
  const float* a = ea + (size_t)e * EDD;
  float* s = sl + (size_t)d * EDD;
#pragma unroll
  for (int j = 0; j < EDD; ++j) atomicAdd(&s[j], a[j]);
  atomicAdd(&deg[d], 1.0f);
}

__global__ void sl_div(float* __restrict__ sl, const float* __restrict__ deg) {
  int i = blockIdx.x * blockDim.x + threadIdx.x;
  if (i >= NN * EDD) return;
  sl[i] /= fmaxf(deg[i >> 3], 1.0f);
}

// ---------------- layer 1 node transform (F_in = 1) ----------------
__global__ void l1_transform(const float* __restrict__ x, const float* __restrict__ Wl,
                             const float* __restrict__ Wr,
                             float* __restrict__ xl, float* __restrict__ xr) {
  int idx = blockIdx.x * blockDim.x + threadIdx.x;
  if (idx >= NN * HCC) return;
  int i = idx >> 7, c = idx & 127;
  float xv = x[i];
  xl[idx] = xv * Wl[c];
  xr[idx] = xv * Wr[c];
}

// ---------------- fused WMMA GEMM pair ----------------
// outa = in @ Wa, outb = in @ Wb  (in:[N,128], W:[128,128])
// A fragments loaded/converted once, used for both weight matrices (64 wmma/block-wave).
// LDS: both weights transposed to f16, 64KB total, K-rotated by (col&7)*16 halves so
// B-fragment reads (16 contiguous halves, 16-aligned start) spread across bank groups.
__global__ __launch_bounds__(128)
void gemm128x2_wmma(const float* __restrict__ in,
                    const float* __restrict__ Wa, const float* __restrict__ Wb,
                    float* __restrict__ outa, float* __restrict__ outb, int nrows) {
  __shared__ _Float16 Wt[2][128 * 128];   // [mat][col*128 + swizzled_k]

  for (int idx = threadIdx.x; idx < 128 * 128; idx += 128) {
    int k = idx >> 7, n = idx & 127;
    int pos = (n << 7) + ((k + ((n & 7) << 4)) & 127);
    Wt[0][pos] = (_Float16)Wa[idx];
    Wt[1][pos] = (_Float16)Wb[idx];
  }
  __syncthreads();

  const int wave    = threadIdx.x >> 5;
  const int lane    = threadIdx.x & 31;
  const int mLocal  = lane & 15;
  const int hi      = lane >> 4;           // 0: lanes 0-15, 1: lanes 16-31
  const int rowBase = blockIdx.x * 64 + wave * 16;

  int m  = rowBase + mLocal;
  int mc = m < nrows ? m : (nrows - 1);    // clamp A-row address (stores guarded)

  v8f acc[2][8];
#pragma unroll
  for (int w = 0; w < 2; ++w)
#pragma unroll
    for (int nt = 0; nt < 8; ++nt) acc[w][nt] = (v8f){0,0,0,0,0,0,0,0};

  const int kBase = hi * 16;   // A-fragment K sub-offset per lane half
  const int kOffB = hi * 16;   // B-fragment K offset per lane half

#pragma unroll
  for (int kc = 0; kc < 4; ++kc) {
    const int k0 = kc * 32;
    // A fragment: contiguous K order (K = k0 + 16*hi + j), same order as the B fragment
    v16h a;
    const float* ap = in + (size_t)mc * HCC + k0 + kBase;
#pragma unroll
    for (int j = 0; j < 8; ++j) {
      a[j]     = (_Float16)ap[j];          // K = k0 + kBase + j
      a[8 + j] = (_Float16)ap[8 + j];      // K = k0 + kBase + 8 + j
    }
#pragma unroll
    for (int w = 0; w < 2; ++w) {
#pragma unroll
      for (int nt = 0; nt < 8; ++nt) {
        const int col = nt * 16 + mLocal;  // B column (N index)
        const int swk = ((k0 + kOffB) + ((col & 7) << 4)) & 127;
        v16h b;
        const _Float16* bp = &Wt[w][(col << 7) + swk];
#pragma unroll
        for (int j = 0; j < 16; ++j) b[j] = bp[j];
        acc[w][nt] = __builtin_amdgcn_wmma_f32_16x16x32_f16(
            false, a, false, b, (short)0, acc[w][nt], false, false);
      }
    }
  }

  // D layout: VGPR r -> M = rowBase + r + 8*hi, N = nt*16 + (lane&15)
#pragma unroll
  for (int nt = 0; nt < 8; ++nt) {
#pragma unroll
    for (int r = 0; r < 8; ++r) {
      int mm = rowBase + r + (hi << 3);
      if (mm < nrows) {
        outa[(size_t)mm * HCC + nt * 16 + mLocal] = acc[0][nt][r];
        outb[(size_t)mm * HCC + nt * 16 + mLocal] = acc[1][nt][r];
      }
    }
  }
}

// ---------------- attention logits: one wave per edge, coalesced ----------------
// lane = channel-within-head slot; head q handled as channel q*32+lane.
__global__ __launch_bounds__(128)
void attn_logits(const float* __restrict__ ea, const float* __restrict__ sl,
                 const int* __restrict__ src, const int* __restrict__ dst,
                 const float* __restrict__ We, const float* __restrict__ att,
                 const float* __restrict__ xl, const float* __restrict__ xr,
                 float* __restrict__ logits, float* __restrict__ nmax) {
  __shared__ float Wes[EDD * HCC];
  __shared__ float atts[HH * CC];
  for (int i = threadIdx.x; i < EDD * HCC; i += blockDim.x) Wes[i] = We[i];
  for (int i = threadIdx.x; i < HH * CC; i += blockDim.x) atts[i] = att[i];
  __syncthreads();

  const int lane = threadIdx.x & 31;
  const int e = blockIdx.x * 4 + (threadIdx.x >> 5);   // wave-uniform edge id
  if (e >= ETOT) return;

  int s, d;
  const float* a;
  if (e < EORIG) { s = src[e]; d = dst[e]; a = ea + (size_t)e * EDD; }
  else           { s = d = e - EORIG;      a = sl + (size_t)s * EDD; }

  float av[EDD];
#pragma unroll
  for (int j = 0; j < EDD; ++j) av[j] = a[j];          // wave-uniform row

  const float* xls = xl + (size_t)s * HCC;
  const float* xrd = xr + (size_t)d * HCC;

  float lg[HH];
#pragma unroll
  for (int q = 0; q < HH; ++q) {
    const int c = q * CC + lane;                       // coalesced channel
    float emb = 0.f;
#pragma unroll
    for (int j = 0; j < EDD; ++j) emb += av[j] * Wes[j * HCC + c];
    float v = xls[c] + xrd[c] + emb;
    v = v > 0.f ? v : 0.2f * v;                        // leaky_relu(0.2)
    lg[q] = v * atts[c];
  }
  // 32-lane reduction per head
#pragma unroll
  for (int off = 16; off > 0; off >>= 1) {
#pragma unroll
    for (int q = 0; q < HH; ++q) lg[q] += __shfl_xor(lg[q], off);
  }
  if (lane < HH) {
    logits[(size_t)e * HH + lane] = lg[lane];
    atomicMaxF32(&nmax[(size_t)d * HH + lane], lg[lane]);
  }
}

// ---------------- segment-softmax: exp + denominator ----------------
__global__ void attn_exp(float* __restrict__ logits, const float* __restrict__ nmax,
                         float* __restrict__ nsum, const int* __restrict__ dst) {
  int idx = blockIdx.x * blockDim.x + threadIdx.x;
  if (idx >= ETOT * HH) return;
  int e = idx >> 2, h = idx & 3;
  int d = (e < EORIG) ? dst[e] : (e - EORIG);
  float ex = __expf(logits[idx] - nmax[(size_t)d * HH + h]);
  logits[idx] = ex;                       // overwrite logit with exp value
  atomicAdd(&nsum[(size_t)d * HH + h], ex);
}

// ---------------- scatter aggregate: one thread per (edge, channel) ----------------
__global__ __launch_bounds__(256)
void aggregate(const float* __restrict__ logits, const float* __restrict__ nsum,
               const float* __restrict__ xl, const int* __restrict__ src,
               const int* __restrict__ dst, float* __restrict__ acc) {
  int idx = blockIdx.x * blockDim.x + threadIdx.x;     // over ETOT*128
  if (idx >= ETOT * HCC) return;
  int e = idx >> 7, c = idx & 127, h = c >> 5;
  int s, d;
  if (e < EORIG) { s = src[e]; d = dst[e]; }
  else           { s = d = e - EORIG; }
  float alpha = logits[(size_t)e * HH + h] / nsum[(size_t)d * HH + h];
  atomicAdd(&acc[(size_t)d * HCC + c], xl[(size_t)s * HCC + c] * alpha);
}

// ---------------- layer finalize ----------------
__global__ void finalize_concat_relu(const float* __restrict__ acc,
                                     const float* __restrict__ b,
                                     float* __restrict__ h) {
  int idx = blockIdx.x * blockDim.x + threadIdx.x;
  if (idx >= NN * HCC) return;
  h[idx] = fmaxf(acc[idx] + b[idx & 127], 0.f);
}

__global__ void finalize_mean(const float* __restrict__ acc,
                              const float* __restrict__ b3,
                              float* __restrict__ h3) {
  int idx = blockIdx.x * blockDim.x + threadIdx.x;
  if (idx >= NN * CC) return;
  int i = idx >> 5, c = idx & 31;
  const float* a = acc + (size_t)i * HCC + c;
  h3[idx] = 0.25f * (a[0] + a[32] + a[64] + a[96]) + b3[c];
}

// ---------------- multi-aggregation (sum/mean/std/min/max/softmax) ----------------
__global__ __launch_bounds__(32)
void multi_aggr(const float* __restrict__ h3, const float* __restrict__ tptr,
                float* __restrict__ g) {
  int b = blockIdx.x;
  int c = threadIdx.x;                 // channel 0..31, one wave per graph
  float t = tptr[0];
  const float* base = h3 + (size_t)b * NPGG * CC;
  float s = 0.f, s2 = 0.f, mn = __builtin_inff(), mx = -__builtin_inff(), thm = -__builtin_inff();
  for (int n = 0; n < NPGG; ++n) {
    float v = base[n * CC + c];
    s += v; s2 += v * v;
    mn = fminf(mn, v); mx = fmaxf(mx, v);
    thm = fmaxf(thm, t * v);
  }
  float exs = 0.f, exh = 0.f;
  for (int n = 0; n < NPGG; ++n) {
    float v = base[n * CC + c];
    float e = __expf(t * v - thm);
    exs += e; exh += e * v;
  }
  float mean = s * (1.0f / NPGG);
  float var  = fmaxf(s2 * (1.0f / NPGG) - mean * mean, 0.f);
  float std  = sqrtf(var + 1e-5f);
  float soft = exh / exs;
  float* go = g + (size_t)b * AGGD;
  go[0 * CC + c] = s;
  go[1 * CC + c] = mean;
  go[2 * CC + c] = std;
  go[3 * CC + c] = mn;
  go[4 * CC + c] = mx;
  go[5 * CC + c] = soft;
}

// ---------------- MLP head ----------------
__global__ void mlp_fc(const float* __restrict__ in, const float* __restrict__ w,
                       const float* __restrict__ bias, float* __restrict__ out,
                       int kdim, int jdim, int do_relu) {
  int b = blockIdx.x, j = threadIdx.x;
  if (j >= jdim) return;
  float acc = bias[j];
  const float* gi = in + (size_t)b * kdim;
  for (int k = 0; k < kdim; ++k) acc += gi[k] * w[(size_t)k * jdim + j];
  if (do_relu) acc = fmaxf(acc, 0.f);
  out[(size_t)b * jdim + j] = acc;
}

// ---------------- host-side launch ----------------
static inline int cdiv(int a, int b) { return (a + b - 1) / b; }

extern "C" void kernel_launch(void* const* d_in, const int* in_sizes, int n_in,
                              void* d_out, int out_size, void* d_ws, size_t ws_size,
                              hipStream_t stream) {
  const float* x         = (const float*)d_in[0];
  const float* edge_attr = (const float*)d_in[1];
  const float* Wl1 = (const float*)d_in[2];
  const float* Wr1 = (const float*)d_in[3];
  const float* We1 = (const float*)d_in[4];
  const float* att1= (const float*)d_in[5];
  const float* b1  = (const float*)d_in[6];
  const float* Wl2 = (const float*)d_in[7];
  const float* Wr2 = (const float*)d_in[8];
  const float* We2 = (const float*)d_in[9];
  const float* att2= (const float*)d_in[10];
  const float* b2  = (const float*)d_in[11];
  const float* Wl3 = (const float*)d_in[12];
  const float* Wr3 = (const float*)d_in[13];
  const float* We3 = (const float*)d_in[14];
  const float* att3= (const float*)d_in[15];
  const float* b3  = (const float*)d_in[16];
  const float* t_soft = (const float*)d_in[17];
  const float* fc1_w = (const float*)d_in[18];
  const float* fc1_b = (const float*)d_in[19];
  const float* fcm_w = (const float*)d_in[20];
  const float* fcm_b = (const float*)d_in[21];
  const float* fc2_w = (const float*)d_in[22];
  const float* fc2_b = (const float*)d_in[23];
  const int* srcp  = (const int*)d_in[24];
  const int* dstp  = (const int*)d_in[25];
  // d_in[26] = batch (nodes are contiguous per graph: node i -> graph i/NPG)

  // workspace carve-up (floats)
  float* w = (float*)d_ws;
  size_t o = 0;
  float* sl     = w + o; o += (size_t)NN * EDD;
  float* deg    = w + o; o += (size_t)NN;
  float* xl     = w + o; o += (size_t)NN * HCC;
  float* xr     = w + o; o += (size_t)NN * HCC;
  float* hbuf   = w + o; o += (size_t)NN * HCC;
  float* acc    = w + o; o += (size_t)NN * HCC;
  float* logits = w + o; o += (size_t)ETOT * HH;
  float* nmax   = w + o; o += (size_t)NN * HH;
  float* nsum   = w + o; o += (size_t)NN * HH;
  float* h3     = w + o; o += (size_t)NN * CC;
  float* g      = w + o; o += (size_t)BB * AGGD;
  float* z1     = w + o; o += (size_t)BB * 256;
  float* z2     = w + o; o += (size_t)BB * 256;

  const float NEG_INF = -__builtin_inff();

  // ---- self-loop edge attrs (mean of incoming) ----
  fill_f32<<<cdiv(NN * EDD, 256), 256, 0, stream>>>(sl, 0.f, NN * EDD);
  fill_f32<<<cdiv(NN, 256), 256, 0, stream>>>(deg, 0.f, NN);
  sl_accum<<<cdiv(EORIG, 128), 128, 0, stream>>>(edge_attr, dstp, sl, deg);
  sl_div<<<cdiv(NN * EDD, 256), 256, 0, stream>>>(sl, deg);

  // per-layer attention pass
  auto run_attention = [&](const float* We, const float* att) {
    fill_f32<<<cdiv(NN * HH, 256), 256, 0, stream>>>(nmax, NEG_INF, NN * HH);
    fill_f32<<<cdiv(NN * HH, 256), 256, 0, stream>>>(nsum, 0.f, NN * HH);
    fill_f32<<<cdiv(NN * HCC, 256), 256, 0, stream>>>(acc, 0.f, NN * HCC);
    attn_logits<<<cdiv(ETOT, 4), 128, 0, stream>>>(
        edge_attr, sl, srcp, dstp, We, att, xl, xr, logits, nmax);
    attn_exp<<<cdiv(ETOT * HH, 256), 256, 0, stream>>>(logits, nmax, nsum, dstp);
    aggregate<<<cdiv(ETOT * HCC, 256), 256, 0, stream>>>(logits, nsum, xl, srcp, dstp, acc);
  };

  // ---- layer 1 (F_in = 1, trivial transform) ----
  l1_transform<<<cdiv(NN * HCC, 256), 256, 0, stream>>>(x, Wl1, Wr1, xl, xr);
  run_attention(We1, att1);
  finalize_concat_relu<<<cdiv(NN * HCC, 256), 256, 0, stream>>>(acc, b1, hbuf);

  // ---- layer 2 (fused WMMA node transforms) ----
  gemm128x2_wmma<<<cdiv(NN, 64), 128, 0, stream>>>(hbuf, Wl2, Wr2, xl, xr, NN);
  run_attention(We2, att2);
  finalize_concat_relu<<<cdiv(NN * HCC, 256), 256, 0, stream>>>(acc, b2, hbuf);

  // ---- layer 3 (concat=False -> head mean) ----
  gemm128x2_wmma<<<cdiv(NN, 64), 128, 0, stream>>>(hbuf, Wl3, Wr3, xl, xr, NN);
  run_attention(We3, att3);
  finalize_mean<<<cdiv(NN * CC, 256), 256, 0, stream>>>(acc, b3, h3);

  // ---- readout + MLP head ----
  multi_aggr<<<BB, 32, 0, stream>>>(h3, t_soft, g);
  mlp_fc<<<BB, 256, 0, stream>>>(g,  fc1_w, fc1_b, z1, AGGD, 256, 1);
  mlp_fc<<<BB, 256, 0, stream>>>(z1, fcm_w, fcm_b, z2, 256, 256, 1);
  mlp_fc<<<BB, OUTD, 0, stream>>>(z2, fc2_w, fc2_b, (float*)d_out, 256, OUTD, 0);
}
